// QParamsLinear_46815143526624
// MI455X (gfx1250) — 8-bit matrix operand order repaired and checked by reading, compiled, not run on hardware
//
#include <hip/hip_runtime.h>
#include <stdint.h>

typedef __attribute__((ext_vector_type(16))) int   v16i;
typedef __attribute__((ext_vector_type(8)))  float v8f;

#define FP8_E4M3_MAX 448.0f

__device__ __forceinline__ unsigned f32_to_e4m3_sw(float x) {
  unsigned s = (__float_as_uint(x) >> 31) << 7;
  float af = fabsf(x);
  if (!(af == af)) return s | 0x7Fu;
  if (af > FP8_E4M3_MAX) af = FP8_E4M3_MAX;
  if (af < 0.015625f) {
    unsigned n = (unsigned)rintf(af * 512.0f);
    return s | n;
  }
  unsigned bits = __float_as_uint(af);
  int e = (int)(bits >> 23) - 127;
  unsigned m = (bits >> 20) & 0x7u;
  unsigned rest = bits & 0xFFFFFu;
  if (rest > 0x80000u || (rest == 0x80000u && (m & 1u))) m++;
  unsigned E = (unsigned)(e + 7);
  if (m == 8u) { m = 0u; E++; }
  unsigned v = (E << 3) | m;
  if (v > 0x7Eu) v = 0x7Eu;
  return s | v;
}

__device__ __forceinline__ uint32_t pack4_e4m3(float a, float b, float c, float d) {
#if __has_builtin(__builtin_amdgcn_cvt_pk_fp8_f32)
  int p = 0;
  p = __builtin_amdgcn_cvt_pk_fp8_f32(a, b, p, false);
  p = __builtin_amdgcn_cvt_pk_fp8_f32(c, d, p, true);
  return (uint32_t)p;
#else
  return f32_to_e4m3_sw(a) | (f32_to_e4m3_sw(b) << 8) |
         (f32_to_e4m3_sw(c) << 16) | (f32_to_e4m3_sw(d) << 24);
#endif
}

__global__ __launch_bounds__(256)
void quant_fp8_kernel(const float* __restrict__ in,
                      const float* __restrict__ scale_ptr,
                      uint32_t* __restrict__ out, int n4) {
  int i = blockIdx.x * blockDim.x + threadIdx.x;
  if (i >= n4) return;
  float inv = scale_ptr ? (1.0f / scale_ptr[0]) : 1.0f;
  float4 v = reinterpret_cast<const float4*>(in)[i];
  float a = fminf(fmaxf(v.x * inv, -FP8_E4M3_MAX), FP8_E4M3_MAX);
  float b = fminf(fmaxf(v.y * inv, -FP8_E4M3_MAX), FP8_E4M3_MAX);
  float c = fminf(fmaxf(v.z * inv, -FP8_E4M3_MAX), FP8_E4M3_MAX);
  float d = fminf(fmaxf(v.w * inv, -FP8_E4M3_MAX), FP8_E4M3_MAX);
  out[i] = pack4_e4m3(a, b, c, d);
}

__global__ __launch_bounds__(256)
void fp8_gemm_wmma_kernel(const uint8_t* __restrict__ Aq,
                          const uint8_t* __restrict__ Bq,
                          const float* __restrict__ wscale,
                          const float* __restrict__ iscale,
                          const float* __restrict__ bias,
                          float* __restrict__ out,
                          int M, int N, int K) {
  const int lane = threadIdx.x & 31;
  const int wave = threadIdx.x >> 5;
  const int h    = lane >> 4;
  const int l16  = lane & 15;
  const int wm   = wave >> 2;
  const int wn   = wave & 3;

  const int m0 = blockIdx.y * 128 + wm * 64;
  const int n0 = blockIdx.x * 256 + wn * 64;

  v8f acc[4][4] = {};

  for (int k0 = 0; k0 < K; k0 += 128) {
    v16i a[4];
    v16i b[4];

#pragma unroll
    for (int i = 0; i < 4; ++i) {
      const uint8_t* ap = Aq + (size_t)(m0 + i * 16 + l16) * K + k0 + h * 16;
#pragma unroll
      for (int c = 0; c < 8; ++c) {
        const int off = (c >> 1) * 32 + (c & 1) * 8;
        uint2 d = *reinterpret_cast<const uint2*>(ap + off);
        a[i][2 * c]     = (int)d.x;
        a[i][2 * c + 1] = (int)d.y;
      }
    }

#pragma unroll
    for (int j = 0; j < 4; ++j) {
      const uint8_t* bp = Bq + (size_t)(n0 + j * 16 + l16) * K + k0 + h * 16;
#pragma unroll
      for (int c = 0; c < 4; ++c) {
        uint4 d = *reinterpret_cast<const uint4*>(bp + c * 32);
        b[j][4 * c]     = (int)d.x;
        b[j][4 * c + 1] = (int)d.y;
        b[j][4 * c + 2] = (int)d.z;
        b[j][4 * c + 3] = (int)d.w;
      }
    }

#pragma unroll
    for (int i = 0; i < 4; ++i) {
#pragma unroll
      for (int j = 0; j < 4; ++j) {
        acc[i][j] = __builtin_amdgcn_wmma_f32_16x16x128_fp8_fp8(
            a[i], b[j], (short)0, acc[i][j], false, false);
      }
    }
  }

  const float sc = iscale[0] * wscale[0];
#pragma unroll
  for (int j = 0; j < 4; ++j) {
    const int n = n0 + j * 16 + l16;
    const float bj = bias[n];
#pragma unroll
    for (int i = 0; i < 4; ++i) {
#pragma unroll
      for (int r = 0; r < 8; ++r) {
        const int m = m0 + i * 16 + h * 8 + r;
        out[(size_t)m * N + n] = acc[i][j][r] * sc + bj;
      }
    }
  }
}

extern "C" void kernel_launch(void* const* d_in, const int* in_sizes, int n_in,
                              void* d_out, int out_size, void* d_ws, size_t ws_size,
                              hipStream_t stream) {
  const float* x      = (const float*)d_in[0];
  const float* w      = (const float*)d_in[1];
  const float* wscale = (const float*)d_in[2];
  const float* iscale = (const float*)d_in[3];
  const float* bias   = (const float*)d_in[4];
  float* out = (float*)d_out;

  const long xN = in_sizes[0];
  const long wN = in_sizes[1];
  const int  N  = in_sizes[4];
  const int  K  = (int)(wN / N);
  const int  M  = (int)(xN / K);

  uint8_t* ws = (uint8_t*)d_ws;
  uint8_t* xq = ws;
  uint8_t* wq = ws + (size_t)M * K;

  {
    int n4 = (int)(xN / 4);
    quant_fp8_kernel<<<(n4 + 255) / 256, 256, 0, stream>>>(
        x, iscale, (uint32_t*)xq, n4);
  }
  {
    int n4 = (int)(wN / 4);
    quant_fp8_kernel<<<(n4 + 255) / 256, 256, 0, stream>>>(
        w, nullptr, (uint32_t*)wq, n4);
  }
  {
    dim3 grid(N / 256, M / 128);
    fp8_gemm_wmma_kernel<<<grid, dim3(256), 0, stream>>>(
        xq, wq, wscale, iscale, bias, out, M, N, K);
  }
}
